// SeerAttnQwen3Attention_30520037605582
// MI455X (gfx1250) — matrix operand order repaired and checked by reading, not compiled, not run on hardware
//
#include <hip/hip_runtime.h>
#include <hip/hip_bf16.h>

#define B_   2
#define S_   1024
#define DM_  2048
#define H_   16
#define HKV_ 4
#define DH_  128
#define G_   4
#define BLK_ 64
#define NB_  16
#define GH_  128

typedef __attribute__((ext_vector_type(16))) __bf16 v16bf;
typedef __attribute__((ext_vector_type(8)))  __bf16 v8bf;
typedef __attribute__((ext_vector_type(8)))  float  v8f;
typedef __attribute__((ext_vector_type(4)))  int    v4i;

union FragBF { v16bf v; v8bf h[2]; };
union H8     { v8bf  v; unsigned short u[8]; };

// ---- gfx1250 async global->LDS copy (ASYNCcnt path), guarded so build never breaks
#if defined(__has_builtin)
#if __has_builtin(__builtin_amdgcn_global_load_async_to_lds_b128)
#define HAVE_ASYNC_LDS 1
#endif
#endif
#ifndef HAVE_ASYNC_LDS
#define HAVE_ASYNC_LDS 0
#endif

#define AS1 __attribute__((address_space(1)))
#define AS3 __attribute__((address_space(3)))

__device__ __forceinline__ void copy16_g2lds(const unsigned short* g, unsigned short* l) {
#if HAVE_ASYNC_LDS
  // signature (from compiler diagnostic): (v4i AS1* src, v4i AS3* dst, imm offset, imm cpol)
  __builtin_amdgcn_global_load_async_to_lds_b128((AS1 v4i*)g, (AS3 v4i*)l, 0, 0);
#else
  *(v8bf*)l = *(const v8bf*)g;
#endif
}
__device__ __forceinline__ void wait_async_lds() {
#if HAVE_ASYNC_LDS
#if __has_builtin(__builtin_amdgcn_s_wait_asynccnt)
  __builtin_amdgcn_s_wait_asynccnt(0);
#else
  asm volatile("s_wait_asynccnt 0x0" ::: "memory");
#endif
#endif
}

__device__ __forceinline__ unsigned short f2bf(float f) {
  unsigned int u = __float_as_uint(f);
  u += 0x7FFFu + ((u >> 16) & 1u);           // round-to-nearest-even
  return (unsigned short)(u >> 16);
}

__device__ __forceinline__ v8f wmma_bf16(v16bf a, v16bf b, v8f c) {
  // D = A(16x32) * B(32x16) + C, f32 accumulate  ->  v_wmma_f32_16x16x32_bf16
  return __builtin_amdgcn_wmma_f32_16x16x32_bf16(false, a, false, b, (short)0, c, false, false);
}

__device__ __forceinline__ float gmax16(float v) {
  v = fmaxf(v, __shfl_xor(v, 1, 32));
  v = fmaxf(v, __shfl_xor(v, 2, 32));
  v = fmaxf(v, __shfl_xor(v, 4, 32));
  v = fmaxf(v, __shfl_xor(v, 8, 32));
  return v;
}
__device__ __forceinline__ float gsum16(float v) {
  v += __shfl_xor(v, 1, 32);
  v += __shfl_xor(v, 2, 32);
  v += __shfl_xor(v, 4, 32);
  v += __shfl_xor(v, 8, 32);
  return v;
}
__device__ __forceinline__ float wsum32(float v) {
#pragma unroll
  for (int off = 16; off > 0; off >>= 1) v += __shfl_xor(v, off, 32);
  return v;
}
__device__ __forceinline__ float wmax32(float v) {
#pragma unroll
  for (int off = 16; off > 0; off >>= 1) v = fmaxf(v, __shfl_xor(v, off, 32));
  return v;
}

// ---------------------------------------------------------------- f32 -> bf16
__global__ void cvt_bf16_kernel(const float* __restrict__ in,
                                unsigned short* __restrict__ out, int n) {
  int i = blockIdx.x * blockDim.x + threadIdx.x;
  if (i < n) out[i] = f2bf(in[i]);
}

// ------------------------------------------------- bf16 GEMM: C = A(MxK) @ B(KxN)
// 128x128 tile, BK=32, 256 threads = 8 waves in 2x4; wave tile 64x32 = 4x2 WMMA.
__global__ __launch_bounds__(256) void gemm_bf16_kernel(
    const unsigned short* __restrict__ A,
    const unsigned short* __restrict__ Bw,
    float* __restrict__ C, int M, int N, int K) {
  __shared__ unsigned short Asl[128][32];   // [m][k]
  __shared__ unsigned short Bsl[128][32];   // transposed: [n][k]
  const int tid  = threadIdx.x;
  const int lane = tid & 31, wave = tid >> 5;
  const int l15  = lane & 15, lh = lane >> 4;
  const int wm   = wave >> 2, wn = wave & 3;
  const int m0   = blockIdx.y * 128, n0 = blockIdx.x * 128;

  const v8f zv = {0.f,0.f,0.f,0.f,0.f,0.f,0.f,0.f};
  v8f acc[4][2];
#pragma unroll
  for (int i = 0; i < 4; ++i)
#pragma unroll
    for (int j = 0; j < 2; ++j) acc[i][j] = zv;

  for (int k0 = 0; k0 < K; k0 += 32) {
    // stage A tile (row-major) via async global->LDS when available
#pragma unroll
    for (int it = 0; it < 2; ++it) {
      int r = (tid >> 2) + it * 64;
      int c = (tid & 3) * 8;
      copy16_g2lds(&A[(size_t)(m0 + r) * K + k0 + c], &Asl[r][c]);
      if (k0 + 32 < K)
        __builtin_prefetch(&A[(size_t)(m0 + r) * K + k0 + 32 + c], 0, 3);
    }
    // stage B tile transposed: Bsl[n][k] = Bw[k][n]  (scatter; cannot be async)
#pragma unroll
    for (int it = 0; it < 2; ++it) {
      int kk = (tid >> 4) + it * 16;
      int nn = (tid & 15) * 8;
      H8 t;
      t.v = *(const v8bf*)&Bw[(size_t)(k0 + kk) * N + n0 + nn];
#pragma unroll
      for (int j = 0; j < 8; ++j) Bsl[nn + j][kk] = t.u[j];
    }
    wait_async_lds();
    __syncthreads();

    v16bf af[4], bfv[2];
#pragma unroll
    for (int mt = 0; mt < 4; ++mt) {
      FragBF f;
      const unsigned short* p = &Asl[wm * 64 + mt * 16 + l15][lh * 8];
      f.h[0] = *(const v8bf*)p;
      f.h[1] = *(const v8bf*)(p + 16);
      af[mt] = f.v;
    }
#pragma unroll
    for (int nt = 0; nt < 2; ++nt) {
      FragBF f;
      const unsigned short* p = &Bsl[wn * 32 + nt * 16 + l15][lh * 8];
      f.h[0] = *(const v8bf*)p;
      f.h[1] = *(const v8bf*)(p + 16);
      bfv[nt] = f.v;
    }
#pragma unroll
    for (int mt = 0; mt < 4; ++mt)
#pragma unroll
      for (int nt = 0; nt < 2; ++nt)
        acc[mt][nt] = wmma_bf16(af[mt], bfv[nt], acc[mt][nt]);
    __syncthreads();
  }

#pragma unroll
  for (int mt = 0; mt < 4; ++mt)
#pragma unroll
    for (int nt = 0; nt < 2; ++nt)
#pragma unroll
      for (int r = 0; r < 8; ++r) {
        int row = m0 + wm * 64 + mt * 16 + r + lh * 8;
        int col = n0 + wn * 32 + nt * 16 + l15;
        C[(size_t)row * N + col] = acc[mt][nt][r];
      }
}

// --------------------------------------------------------- RMS norm (DH=128)
__global__ __launch_bounds__(128) void rmsnorm_kernel(float* __restrict__ x,
                                                      const float* __restrict__ w) {
  int row = blockIdx.x, d = threadIdx.x;
  __shared__ float red[128];
  float v = x[(size_t)row * DH_ + d];
  red[d] = v * v;
  __syncthreads();
  for (int s2 = 64; s2 > 0; s2 >>= 1) {
    if (d < s2) red[d] += red[d + s2];
    __syncthreads();
  }
  float rr = rsqrtf(red[0] * (1.0f / DH_) + 1e-6f);
  x[(size_t)row * DH_ + d] = w[d] * (v * rr);
}

// --------------------------------- qg = mean over G of q_norm, then RoPE(gq)
__global__ __launch_bounds__(128) void qg_rope_kernel(const float* __restrict__ q,
                                                      const float* __restrict__ cg,
                                                      const float* __restrict__ sg,
                                                      float* __restrict__ qg) {
  int row = blockIdx.x;                 // (b*S+s)*HKV + hkv
  int d = threadIdx.x;
  int hkv = row % HKV_;
  int bs = row / HKV_;
  int s = bs % S_;
  __shared__ float sh[DH_];
  float acc = 0.f;
#pragma unroll
  for (int g = 0; g < G_; ++g)
    acc += q[((size_t)bs * H_ + hkv * G_ + g) * DH_ + d];
  acc *= 0.25f;
  sh[d] = acc;
  __syncthreads();
  float rot = (d < 64) ? -sh[d + 64] : sh[d - 64];
  qg[(size_t)row * DH_ + d] = acc * cg[s * DH_ + d] + rot * sg[s * DH_ + d];
}

// ---------------------------- kp = mean over block of k_norm, then RoPE(blk)
__global__ __launch_bounds__(128) void kp_rope_kernel(const float* __restrict__ k,
                                                      const float* __restrict__ cb,
                                                      const float* __restrict__ sb,
                                                      float* __restrict__ kp) {
  int row = blockIdx.x;                 // (b*NB+nb)*HKV + hkv
  int d = threadIdx.x;
  int hkv = row % HKV_;
  int bn = row / HKV_;
  int nb = bn % NB_;
  int b = bn / NB_;
  __shared__ float sh[DH_];
  float acc = 0.f;
  for (int j = 0; j < BLK_; ++j)
    acc += k[(((size_t)(b * S_ + nb * BLK_ + j)) * HKV_ + hkv) * DH_ + d];
  acc *= (1.0f / BLK_);
  sh[d] = acc;
  __syncthreads();
  float rot = (d < 64) ? -sh[d + 64] : sh[d - 64];
  kp[(size_t)row * DH_ + d] = acc * cb[nb * DH_ + d] + rot * sb[nb * DH_ + d];
}

// ------------------------------------------ RoPE + convert to bf16 (q and k)
__global__ __launch_bounds__(128) void rope_cvt_kernel(const float* __restrict__ x,
                                                       const float* __restrict__ cv,
                                                       const float* __restrict__ sv,
                                                       unsigned short* __restrict__ out,
                                                       int nh) {
  int row = blockIdx.x;                 // (b*S+s)*nh + head
  int d = threadIdx.x;
  int s = (row / nh) % S_;
  size_t base = (size_t)row * DH_;
  float v = x[base + d];
  float rot = (d < 64) ? -x[base + d + 64] : x[base + d - 64];
  out[base + d] = f2bf(v * cv[s * DH_ + d] + rot * sv[s * DH_ + d]);
}

// --------------------------------------------------- kgp = kp @ gate_wk (tiny)
__global__ __launch_bounds__(128) void kgp_kernel(const float* __restrict__ kp,
                                                  const float* __restrict__ wkg,
                                                  float* __restrict__ kgp) {
  int row = blockIdx.x, e = threadIdx.x;
  float acc = 0.f;
  for (int d = 0; d < DH_; ++d) acc += kp[(size_t)row * DH_ + d] * wkg[d * GH_ + e];
  kgp[(size_t)row * GH_ + e] = acc;
}

// ------------------------- gate scores = (qg @ gate_wq) . kgp, masked (b,hkv,s,n)
__global__ __launch_bounds__(128) void gate_scores_kernel(const float* __restrict__ qg,
                                                          const float* __restrict__ wqg,
                                                          const float* __restrict__ kgp,
                                                          float* __restrict__ sco) {
  int blk = blockIdx.x;                 // (b*HKV+hkv)*S + s
  int s = blk % S_;
  int bh = blk / S_;
  int hkv = bh % HKV_;
  int b = bh / HKV_;
  int e = threadIdx.x;
  __shared__ float qp[GH_];
  size_t qrow = (size_t)(b * S_ + s) * HKV_ + hkv;
  float acc = 0.f;
  for (int d = 0; d < DH_; ++d) acc += qg[qrow * DH_ + d] * wqg[d * GH_ + e];
  qp[e] = acc;
  __syncthreads();
  if (e < NB_) {
    int n = e;
    size_t kr = (size_t)(b * NB_ + n) * HKV_ + hkv;
    float sc = 0.f;
    for (int i = 0; i < GH_; ++i) sc += qp[i] * kgp[kr * GH_ + i];
    sc *= 0.08838834764831845f;         // 1/sqrt(GH)
    if (n * BLK_ > s) sc = -3.4028235e38f;
    sco[(size_t)blk * NB_ + n] = sc;
  }
}

// ------------------------------------------- flash attention, 64-row q tiles
// grid (S/64, B*H), 128 thr = 4 waves; wave w owns score rows w*16..w*16+15.
__global__ __launch_bounds__(128) void attn_kernel(
    const unsigned short* __restrict__ Q,   // (B,S,H,DH) bf16 roped
    const unsigned short* __restrict__ Kb,  // (B,S,HKV,DH) bf16 roped
    const unsigned short* __restrict__ Vb,  // (B,S,HKV,DH) bf16
    float* __restrict__ Ao,                 // (B,S,H,DH) f32
    float* __restrict__ Gh) {               // (B,H,S,NB) f32 blockwise P-max
  __shared__ unsigned short Kl[64][128];    // [key][dh]
  __shared__ unsigned short Vt[128][64];    // [dh][key]  (transposed)
  __shared__ unsigned short Pl[64][64];     // P as bf16
  __shared__ float Mb[64][NB_];             // per-row per-block raw score max
  const int tid = threadIdx.x, lane = tid & 31, w = tid >> 5;
  const int l15 = lane & 15, lh = lane >> 4;
  const int qt = blockIdx.x, bh = blockIdx.y;
  const int b = bh / H_, h = bh % H_, hkv = h / G_;
  const int q0 = qt * 64;

  // Q fragments stay in VGPRs for the whole tile
  v16bf qa[4];
  {
    int qrow = q0 + w * 16 + l15;
    const unsigned short* qp = Q + (((size_t)(b * S_ + qrow)) * H_ + h) * DH_;
#pragma unroll
    for (int kk = 0; kk < 4; ++kk) {
      FragBF f;
      f.h[0] = *(const v8bf*)(qp + kk * 32 + lh * 8);
      f.h[1] = *(const v8bf*)(qp + kk * 32 + lh * 8 + 16);
      qa[kk] = f.v;
    }
  }

  const v8f zv = {0.f,0.f,0.f,0.f,0.f,0.f,0.f,0.f};
  v8f o[8];
#pragma unroll
  for (int dt = 0; dt < 8; ++dt) o[dt] = zv;
  float mrow[8], lrow[8];
#pragma unroll
  for (int r = 0; r < 8; ++r) { mrow[r] = -3.0e38f; lrow[r] = 0.f; }

  for (int n = 0; n <= qt; ++n) {
    __syncthreads();
    // stage K block (row-major: dh contiguous) -- async global->LDS when available
    {
      int j = tid >> 1, d0 = (tid & 1) * 64;
      const unsigned short* kp = Kb + (((size_t)(b * S_ + n * 64 + j)) * HKV_ + hkv) * DH_ + d0;
#pragma unroll
      for (int i = 0; i < 8; ++i)
        copy16_g2lds(kp + i * 8, &Kl[j][d0 + i * 8]);
    }
    // stage V transposed (scatter; cannot be async)
#pragma unroll
    for (int it = 0; it < 8; ++it) {
      int j = (tid >> 4) + it * 8;
      int d0 = (tid & 15) * 8;
      H8 t;
      t.v = *(const v8bf*)(Vb + (((size_t)(b * S_ + n * 64 + j)) * HKV_ + hkv) * DH_ + d0);
#pragma unroll
      for (int i = 0; i < 8; ++i) Vt[d0 + i][j] = t.u[i];
    }
    wait_async_lds();
    __syncthreads();

    // S = Q @ K^T  (4 n-tiles x 4 k-chunks)
    v8f sc[4];
#pragma unroll
    for (int nt = 0; nt < 4; ++nt) sc[nt] = zv;
#pragma unroll
    for (int kk = 0; kk < 4; ++kk) {
#pragma unroll
      for (int nt = 0; nt < 4; ++nt) {
        FragBF f;
        const unsigned short* p = &Kl[nt * 16 + l15][kk * 32 + lh * 8];
        f.h[0] = *(const v8bf*)p;
        f.h[1] = *(const v8bf*)(p + 16);
        sc[nt] = wmma_bf16(qa[kk], f.v, sc[nt]);
      }
    }

    // online softmax update + record block max + write P(bf16) to LDS
    const float scale = 0.08838834764831845f;   // 1/sqrt(DH)
#pragma unroll
    for (int r = 0; r < 8; ++r) {
      int rowl = w * 16 + r + lh * 8;
      int rowg = q0 + rowl;
      float pv[4];
      float bm = -3.0e38f;
#pragma unroll
      for (int nt = 0; nt < 4; ++nt) {
        float v = sc[nt][r] * scale;
        int colg = n * 64 + nt * 16 + l15;
        if (colg > rowg) v = -3.0e38f;
        pv[nt] = v;
        bm = fmaxf(bm, v);
      }
      bm = gmax16(bm);
      if (l15 == 0) Mb[rowl][n] = bm;
      float mn = fmaxf(mrow[r], bm);
      float al = __expf(mrow[r] - mn);
      float rs = 0.f;
#pragma unroll
      for (int nt = 0; nt < 4; ++nt) {
        float p = __expf(pv[nt] - mn);
        rs += p;
        Pl[rowl][nt * 16 + l15] = f2bf(p);
      }
      rs = gsum16(rs);
      lrow[r] = lrow[r] * al + rs;
      mrow[r] = mn;
#pragma unroll
      for (int dt = 0; dt < 8; ++dt) o[dt][r] = o[dt][r] * al;
    }
    __syncthreads();

    // O += P @ V   (8 dh-tiles x 2 key-chunks)
#pragma unroll
    for (int kc = 0; kc < 2; ++kc) {
      FragBF pa;
      const unsigned short* pp = &Pl[w * 16 + l15][kc * 32 + lh * 8];
      pa.h[0] = *(const v8bf*)pp;
      pa.h[1] = *(const v8bf*)(pp + 16);
      v16bf pav = pa.v;
#pragma unroll
      for (int dt = 0; dt < 8; ++dt) {
        FragBF vf;
        const unsigned short* vp = &Vt[dt * 16 + l15][kc * 32 + lh * 8];
        vf.h[0] = *(const v8bf*)vp;
        vf.h[1] = *(const v8bf*)(vp + 16);
        o[dt] = wmma_bf16(pav, vf.v, o[dt]);
      }
    }
  }
  __syncthreads();

  // epilogue: O/l and gt = exp(m_blk - m)/l per 64-block
#pragma unroll
  for (int r = 0; r < 8; ++r) {
    int rowl = w * 16 + r + lh * 8;
    int rowg = q0 + rowl;
    float inv = 1.0f / lrow[r];
#pragma unroll
    for (int dt = 0; dt < 8; ++dt)
      Ao[(((size_t)(b * S_ + rowg)) * H_ + h) * DH_ + dt * 16 + l15] = o[dt][r] * inv;
    int nn = l15;
    float g = 0.f;
    if (nn <= qt) g = __expf(Mb[rowl][nn] - mrow[r]) * inv;
    Gh[(((size_t)(b * H_ + h)) * S_ + rowg) * NB_ + nn] = g;
  }
}

// ------------------------------------------------------------- loss reduction
__global__ void zero_kernel(float* p) { if (threadIdx.x == 0 && blockIdx.x == 0) *p = 0.f; }

__global__ __launch_bounds__(32) void loss_kernel(const float* __restrict__ Gh,
                                                  const float* __restrict__ sco,
                                                  float* __restrict__ loss) {
  int blk = blockIdx.x;                       // (b*HKV+hkv)*(S/2) + (s-512)
  int s = (S_ / 2) + (blk % (S_ / 2));
  int bh = blk / (S_ / 2);
  int hkv = bh % HKV_;
  int b = bh / HKV_;
  int n = threadIdx.x;                        // lanes 16..31 are padding
  float g = 0.f;
  if (n < NB_) {
#pragma unroll
    for (int gg = 0; gg < G_; ++gg) {
      float v = Gh[(((size_t)(b * H_ + hkv * G_ + gg)) * S_ + s) * NB_ + n];
      g = fmaxf(g, v);
    }
    if (n * BLK_ > s) g = 0.f;
  }
  float tsum = wsum32(g);
  float t = g / (tsum + 1e-9f);
  float sc = (n < NB_) ? sco[(((size_t)(b * HKV_ + hkv)) * S_ + s) * NB_ + n]
                       : -3.4028235e38f;
  float mx = wmax32(sc);
  float ex = __expf(sc - mx);
  float es = wsum32(ex);
  float lse = mx + __logf(es);
  float kl = 0.f;
  if (n < NB_ && t > 0.f) kl = t * (__logf(t) - (sc - lse));
  kl = wsum32(kl);
  if (threadIdx.x == 0)
    atomicAdd(loss, kl * (1.0f / (B_ * HKV_ * (S_ / 2) * NB_)));
}

// =============================================================== host driver
extern "C" void kernel_launch(void* const* d_in, const int* in_sizes, int n_in,
                              void* d_out, int out_size, void* d_ws, size_t ws_size,
                              hipStream_t stream) {
  (void)in_sizes; (void)n_in; (void)out_size; (void)ws_size;
  const float* hidden = (const float*)d_in[0];
  const float* wq  = (const float*)d_in[1];
  const float* wk  = (const float*)d_in[2];
  const float* wv  = (const float*)d_in[3];
  const float* wo  = (const float*)d_in[4];
  const float* qnw = (const float*)d_in[5];
  const float* knw = (const float*)d_in[6];
  const float* gwq = (const float*)d_in[7];
  const float* gwk = (const float*)d_in[8];
  const float* cosv = (const float*)d_in[9];
  const float* sinv = (const float*)d_in[10];
  const float* cosg = (const float*)d_in[11];
  const float* sing = (const float*)d_in[12];
  const float* cosb = (const float*)d_in[13];
  const float* sinb = (const float*)d_in[14];
  // d_in[15] (block_attention_mask) is deterministic: n*BLK <= s, computed analytically

  char* ws = (char*)d_ws;
  unsigned short* hb    = (unsigned short*)(ws + 0);                       // 8 MB
  unsigned short* wslot = (unsigned short*)(ws + (size_t)(8)  * 1048576);  // 8 MB (reused per weight)
  float* qf   = (float*)(ws + (size_t)(16) * 1048576);                     // 16 MB (reused: attn out f32)
  float* kf   = (float*)(ws + (size_t)(32) * 1048576);                     // 4 MB
  float* vf   = (float*)(ws + (size_t)(36) * 1048576);                     // 4 MB (reused: gt_h)
  float* qg   = (float*)(ws + (size_t)(40) * 1048576);                     // 4 MB
  float* kp   = (float*)(ws + (size_t)(44) * 1048576);                     // 64 KB
  float* kgp  = (float*)(ws + (size_t)(44) * 1048576 + 262144);            // 64 KB
  float* sco  = (float*)(ws + (size_t)(45) * 1048576);                     // 512 KB
  unsigned short* qrb = (unsigned short*)(ws + (size_t)(46) * 1048576);    // 8 MB
  unsigned short* krb = (unsigned short*)(ws + (size_t)(54) * 1048576);    // 2 MB
  unsigned short* vb  = (unsigned short*)(ws + (size_t)(57) * 1048576);    // 2 MB
  unsigned short* ab  = (unsigned short*)(ws + (size_t)(60) * 1048576);    // 8 MB -> 68 MB total
  float* attnf = qf;
  float* gth   = vf;
  float* outp  = (float*)d_out;
  float* loss  = outp + (size_t)B_ * S_ * DM_;

  const int nHid = B_ * S_ * DM_;           // 4 M
  const int nWq  = DM_ * H_ * DH_;          // 4 M
  const int nWkv = DM_ * HKV_ * DH_;        // 1 M
  const int nQ   = B_ * S_ * H_ * DH_;      // 4 M
  const int nK   = B_ * S_ * HKV_ * DH_;    // 1 M

  // hidden -> bf16
  cvt_bf16_kernel<<<(nHid + 255) / 256, 256, 0, stream>>>(hidden, hb, nHid);

  // q = X @ Wq
  cvt_bf16_kernel<<<(nWq + 255) / 256, 256, 0, stream>>>(wq, wslot, nWq);
  gemm_bf16_kernel<<<dim3((H_ * DH_) / 128, (B_ * S_) / 128), 256, 0, stream>>>(
      hb, wslot, qf, B_ * S_, H_ * DH_, DM_);
  // k = X @ Wk
  cvt_bf16_kernel<<<(nWkv + 255) / 256, 256, 0, stream>>>(wk, wslot, nWkv);
  gemm_bf16_kernel<<<dim3((HKV_ * DH_) / 128, (B_ * S_) / 128), 256, 0, stream>>>(
      hb, wslot, kf, B_ * S_, HKV_ * DH_, DM_);
  // v = X @ Wv
  cvt_bf16_kernel<<<(nWkv + 255) / 256, 256, 0, stream>>>(wv, wslot, nWkv);
  gemm_bf16_kernel<<<dim3((HKV_ * DH_) / 128, (B_ * S_) / 128), 256, 0, stream>>>(
      hb, wslot, vf, B_ * S_, HKV_ * DH_, DM_);

  // RMS norm q, k (in place)
  rmsnorm_kernel<<<B_ * S_ * H_, 128, 0, stream>>>(qf, qnw);
  rmsnorm_kernel<<<B_ * S_ * HKV_, 128, 0, stream>>>(kf, knw);

  // v -> bf16 (no rope)
  cvt_bf16_kernel<<<(nK + 255) / 256, 256, 0, stream>>>(vf, vb, nK);

  // gate branches
  qg_rope_kernel<<<B_ * S_ * HKV_, 128, 0, stream>>>(qf, cosg, sing, qg);
  kp_rope_kernel<<<B_ * NB_ * HKV_, 128, 0, stream>>>(kf, cosb, sinb, kp);
  kgp_kernel<<<B_ * NB_ * HKV_, 128, 0, stream>>>(kp, gwk, kgp);
  gate_scores_kernel<<<B_ * HKV_ * S_, 128, 0, stream>>>(qg, gwq, kgp, sco);

  // RoPE + bf16 for attention
  rope_cvt_kernel<<<B_ * S_ * H_, 128, 0, stream>>>(qf, cosv, sinv, qrb, H_);
  rope_cvt_kernel<<<B_ * S_ * HKV_, 128, 0, stream>>>(kf, cosv, sinv, krb, HKV_);

  // flash attention (+ blockwise P-max for gate target)
  attn_kernel<<<dim3(S_ / 64, B_ * H_), 128, 0, stream>>>(qrb, krb, vb, attnf, gth);

  // output projection -> d_out
  cvt_bf16_kernel<<<(nQ + 255) / 256, 256, 0, stream>>>(attnf, ab, nQ);
  cvt_bf16_kernel<<<(nWq + 255) / 256, 256, 0, stream>>>(wo, wslot, nWq);
  gemm_bf16_kernel<<<dim3(DM_ / 128, (B_ * S_) / 128), 256, 0, stream>>>(
      ab, wslot, outp, B_ * S_, DM_, H_ * DH_);

  // gate loss
  zero_kernel<<<1, 1, 0, stream>>>(loss);
  loss_kernel<<<B_ * HKV_ * (S_ / 2), 32, 0, stream>>>(gth, sco, loss);
}
